// WeaklySupervisedAttention_68564857914106
// MI455X (gfx1250) — matrix operand order repaired and checked by reading, not compiled, not run on hardware
//
#include <hip/hip_runtime.h>
#include <math.h>

// ---------------------------------------------------------------------------
// WeaklySupervisedAttention for MI455X (gfx1250, wave32, WMMA bf16)
//   q = query @ Wq + bq                      [B,SQ,U]  (bf16 ws)
//   k = value @ Wk + bk + weak_labels        [B,SK,U]  (bf16 ws)
//   v = value @ Wv + bv                      [B,SK,U]  (bf16 ws)
//   S = q @ k^T   -> softmax -> P            (f32 / bf16 ws)
//   out = P @ v                              [B,SQ,U]  (f32)
// GEMMs: v_wmma_f32_16x16x32_bf16, 128x128x32 tiles, 8 waves/block,
// double-buffered LDS; bf16 tiles staged by the Tensor Data Mover
// (tensor_load_to_lds + s_wait_tensorcnt); f32 operands converted on the
// VALU (v_cvt_pk_bf16_f32 if available). B-fragment loads are
// software-pipelined so WMMAs issue back-to-back.
// ---------------------------------------------------------------------------

typedef __attribute__((ext_vector_type(16))) __bf16 v16bf;
typedef __attribute__((ext_vector_type(8)))  float  v8f;
typedef unsigned short     u16t;
typedef unsigned int       u32t;
typedef unsigned long long u64t;
typedef __attribute__((ext_vector_type(4))) unsigned int u32x4;
typedef __attribute__((ext_vector_type(8))) int          i32x8;
typedef __attribute__((ext_vector_type(4))) int          i32x4;

#if defined(__has_builtin)
#if __has_builtin(__builtin_amdgcn_tensor_load_to_lds)
#define WSATTN_HAS_TDM 1
#endif
#if __has_builtin(__builtin_amdgcn_cvt_pk_bf16_f32)
#define WSATTN_HAS_PKCVT 1
#endif
#endif
#ifndef WSATTN_HAS_TDM
#define WSATTN_HAS_TDM 0
#endif
#ifndef WSATTN_HAS_PKCVT
#define WSATTN_HAS_PKCVT 0
#endif

__device__ __forceinline__ u16t f2bf(float x) {
    u32t u = __float_as_uint(x);
    u32t lsb = (u >> 16) & 1u;           // round-to-nearest-even
    u += 0x7fffu + lsb;
    return (u16t)(u >> 16);
}

__device__ __forceinline__ u32t pack2bf(float x, float y) {
#if WSATTN_HAS_PKCVT
    typedef __attribute__((ext_vector_type(2))) __bf16 v2bf;
    v2bf r = __builtin_amdgcn_cvt_pk_bf16_f32(x, y);
    return __builtin_bit_cast(u32t, r);
#else
    return (u32t)f2bf(x) | ((u32t)f2bf(y) << 16);
#endif
}

#if WSATTN_HAS_TDM
// One-shot TDM: DMA a tile_m x tile_k (rows x K, 2-byte elems, row stride
// stride_k elems) tile from global into LDS at byte offset lds_off.
// D# layout per CDNA5 ISA ch.8 (group0: flags/lds/global addr; group1: dims).
// This toolchain exposes the 6-arg builtin:
//   (uint32x4 g0, int32x8 g1, int32x4 g2, int32x4 g3, int32x8 pad, i32 cpol)
__device__ __forceinline__ void tdm_load_2d(const void* gaddr, u32t lds_off,
                                            u32t tile_k, u32t tile_m,
                                            u32t tensor_k, u32t stride_k) {
    u64t ga = (u64t)gaddr;
    u32x4 g0;
    g0[0] = 0x1u;                                  // count=1, user mode
    g0[1] = lds_off;                               // lds_addr
    g0[2] = (u32t)ga;                              // global_addr[31:0]
    g0[3] = ((u32t)(ga >> 32) & 0x01FFFFFFu)       // global_addr[56:32]
          | 0x80000000u;                           // type=2 ("image")
    i32x8 g1;
    g1[0] = (int)(1u << 16);                       // data_size=2B, wg_mask=0
    g1[1] = (int)((tensor_k & 0xFFFFu) << 16);     // tensor_dim0[15:0]
    g1[2] = (int)(((tensor_k >> 16) & 0xFFFFu)     // tensor_dim0[31:16]
          | ((0x100000u & 0xFFFFu) << 16));        // tensor_dim1[15:0] (1<<20)
    g1[3] = (int)(((0x100000u >> 16) & 0xFFFFu)    // tensor_dim1[31:16]
          | ((tile_k & 0xFFFFu) << 16));           // tile_dim0
    g1[4] = (int)(tile_m & 0xFFFFu);               // tile_dim1 (tile_dim2=0)
    g1[5] = (int)stride_k;                         // tensor_dim0_stride[31:0]
    g1[6] = 0;
    g1[7] = 0;
    i32x4 z4 = {0, 0, 0, 0};
    i32x8 z8 = {0, 0, 0, 0, 0, 0, 0, 0};
    __builtin_amdgcn_tensor_load_to_lds(g0, g1, z4, z4, z8, 0);
}
#else
// Fallback: per-lane async DMA global -> LDS, 16B/lane, ASYNCcnt-tracked.
__device__ __forceinline__ void async_ld_b128(u16t* ldsDst, const u16t* gsrc) {
    u32t lds = (u32t)(size_t)ldsDst;
    asm volatile("global_load_async_to_lds_b128 %0, %1, off"
                 :: "v"(lds), "v"(gsrc) : "memory");
}
__device__ __forceinline__ void wait_async0() {
    asm volatile("s_wait_asynccnt 0x0" ::: "memory");
}
#endif

union Frag { uint4 q[2]; v16bf v; };

// AKIND: 0 = f32 global A (M,K); 1 = bf16 ws A (M,K) -> TDM/async staged
// BKIND: 0 = f32 (K,N) weights (convert+transpose via VALU)
//        1 = bf16 (N,K) k-matrix (K-contiguous -> TDM/async staged)
//        2 = bf16 (K,N) v-matrix (transpose via VALU)
template <int AKIND, int BKIND, bool OUTBF, bool HAS_BIAS, bool HAS_ADDEND>
__global__ __launch_bounds__(256)
void wsattn_gemm_wmma(const void* __restrict__ Aptr, int lda, u64t strideA,
                      const void* __restrict__ Bptr, int ldb, u64t strideB,
                      const float* __restrict__ bias,
                      const float* __restrict__ addend, int ldadd, u64t strideAdd,
                      void* __restrict__ Cptr, int ldc, u64t strideC, int K) {
    constexpr int BK = 32;
    __shared__ __align__(16) u16t As[2][128 * BK];   // [m][k]
    __shared__ __align__(16) u16t Bs[2][128 * BK];   // [n][k] K-contiguous

    const int tid  = threadIdx.x;
    const int wave = tid >> 5;
    const int lane = tid & 31;
    const int lh   = lane >> 4;
    const int l16  = lane & 15;
    const u64t bz  = blockIdx.z;

    const int rowBase = blockIdx.y * 128;
    const int colBase = blockIdx.x * 128;

    const float* Af = (const float*)Aptr + bz * strideA;
    const u16t*  Ab = (const u16t*)Aptr  + bz * strideA;
    const float* Bf = (const float*)Bptr + bz * strideB;
    const u16t*  Bb = (const u16t*)Bptr  + bz * strideB;
    const float* Ad = HAS_ADDEND ? addend + bz * strideAdd : nullptr;

    v8f acc[8] = {};

    // ---- staging helpers -------------------------------------------------
    auto stageA_bf16 = [&](int buf, int k0) {      // AKIND==1
#if WSATTN_HAS_TDM
        if (wave == 0)
            tdm_load_2d(Ab + (size_t)rowBase * lda + k0,
                        (u32t)(size_t)&As[buf][0], BK, 128, (u32t)lda, (u32t)lda);
#else
        #pragma unroll
        for (int j = 0; j < 2; ++j) {
            int c = tid + 256 * j, m = c >> 2, kq = c & 3;
            async_ld_b128(&As[buf][m * BK + kq * 8],
                          Ab + (size_t)(rowBase + m) * lda + k0 + kq * 8);
        }
#endif
    };
    auto stageB_bf16 = [&](int buf, int k0) {      // BKIND==1
#if WSATTN_HAS_TDM
        if (wave == 0)
            tdm_load_2d(Bb + (size_t)colBase * ldb + k0,
                        (u32t)(size_t)&Bs[buf][0], BK, 128, (u32t)ldb, (u32t)ldb);
#else
        #pragma unroll
        for (int j = 0; j < 2; ++j) {
            int c = tid + 256 * j, n = c >> 2, kq = c & 3;
            async_ld_b128(&Bs[buf][n * BK + kq * 8],
                          Bb + (size_t)(colBase + n) * ldb + k0 + kq * 8);
        }
#endif
    };
    auto drain_dma = [&]() {
#if WSATTN_HAS_TDM
        if (wave == 0) __builtin_amdgcn_s_wait_tensorcnt(0);
#else
        wait_async0();
#endif
    };
    auto loadA_f32 = [&](int k0, float4* r) {      // AKIND==0
        #pragma unroll
        for (int j = 0; j < 4; ++j) {
            int c = tid + 256 * j, m = c >> 3, kq = c & 7;
            r[j] = *(const float4*)(Af + (size_t)(rowBase + m) * lda + k0 + kq * 4);
        }
    };
    auto storeA_f32 = [&](int buf, const float4* r) {
        #pragma unroll
        for (int j = 0; j < 4; ++j) {
            int c = tid + 256 * j, m = c >> 3, kq = c & 7;
            *(uint2*)(&As[buf][m * BK + kq * 4]) =
                make_uint2(pack2bf(r[j].x, r[j].y), pack2bf(r[j].z, r[j].w));
        }
    };
    auto loadB_f32 = [&](int k0, float4* r) {      // BKIND==0
        #pragma unroll
        for (int j = 0; j < 4; ++j) {
            int c = tid + 256 * j, k = c >> 5, nq = c & 31;
            r[j] = *(const float4*)(Bf + (size_t)(k0 + k) * ldb + colBase + nq * 4);
        }
    };
    auto storeB_f32 = [&](int buf, const float4* r) {
        #pragma unroll
        for (int j = 0; j < 4; ++j) {
            int c = tid + 256 * j, k = c >> 5, nq = c & 31;
            Bs[buf][(nq * 4 + 0) * BK + k] = f2bf(r[j].x);
            Bs[buf][(nq * 4 + 1) * BK + k] = f2bf(r[j].y);
            Bs[buf][(nq * 4 + 2) * BK + k] = f2bf(r[j].z);
            Bs[buf][(nq * 4 + 3) * BK + k] = f2bf(r[j].w);
        }
    };
    auto loadB_kn16 = [&](int k0, uint2* r) {      // BKIND==2
        #pragma unroll
        for (int j = 0; j < 4; ++j) {
            int c = tid + 256 * j, k = c >> 5, nq = c & 31;
            r[j] = *(const uint2*)(Bb + (size_t)(k0 + k) * ldb + colBase + nq * 4);
        }
    };
    auto storeB_kn16 = [&](int buf, const uint2* r) {
        #pragma unroll
        for (int j = 0; j < 4; ++j) {
            int c = tid + 256 * j, k = c >> 5, nq = c & 31;
            Bs[buf][(nq * 4 + 0) * BK + k] = (u16t)(r[j].x);
            Bs[buf][(nq * 4 + 1) * BK + k] = (u16t)(r[j].x >> 16);
            Bs[buf][(nq * 4 + 2) * BK + k] = (u16t)(r[j].y);
            Bs[buf][(nq * 4 + 3) * BK + k] = (u16t)(r[j].y >> 16);
        }
    };

    // ---- prologue: stage tile 0 into buffer 0 ----------------------------
    {
        if constexpr (AKIND == 0) { float4 r[4]; loadA_f32(0, r); storeA_f32(0, r); }
        else                      { stageA_bf16(0, 0); }
        if constexpr (BKIND == 0)      { float4 r[4]; loadB_f32(0, r); storeB_f32(0, r); }
        else if constexpr (BKIND == 2) { uint2 r[4]; loadB_kn16(0, r); storeB_kn16(0, r); }
        else                           { stageB_bf16(0, 0); }
        if constexpr (AKIND == 1 || BKIND == 1) drain_dma();
        __syncthreads();
    }

    const int nk = K / BK;
    for (int kt = 0; kt < nk; ++kt) {
        const int p = kt & 1, qb = p ^ 1;
        const bool haveNext = (kt + 1) < nk;
        const int knext = (kt + 1) * BK;

        // prefetch next tile (issue DMA / global loads early)
        float4 ra[4]; float4 rb[4]; uint2 rb2[4];
        if (haveNext) {
            if constexpr (AKIND == 0) loadA_f32(knext, ra);
            else                      stageA_bf16(qb, knext);
            if constexpr (BKIND == 0)      loadB_f32(knext, rb);
            else if constexpr (BKIND == 2) loadB_kn16(knext, rb2);
            else                           stageB_bf16(qb, knext);
        }

        // ---- compute from buffer p: 1 A frag + 8 pipelined B frags ------
        {
            const u16t* Ap = As[p];
            const u16t* Bp = Bs[p];
            Frag afr;
            {
                const u16t* ap = &Ap[(wave * 16 + l16) * BK + lh * 16];
                afr.q[0] = *(const uint4*)(ap);
                afr.q[1] = *(const uint4*)(ap + 8);
            }
            Frag bfr[2];
            {
                const u16t* bp = &Bp[l16 * BK + lh * 16];
                bfr[0].q[0] = *(const uint4*)(bp);
                bfr[0].q[1] = *(const uint4*)(bp + 8);
            }
            #pragma unroll
            for (int nt = 0; nt < 8; ++nt) {
                if (nt < 7) {
                    const u16t* bp = &Bp[((nt + 1) * 16 + l16) * BK + lh * 16];
                    bfr[(nt + 1) & 1].q[0] = *(const uint4*)(bp);
                    bfr[(nt + 1) & 1].q[1] = *(const uint4*)(bp + 8);
                }
                acc[nt] = __builtin_amdgcn_wmma_f32_16x16x32_bf16(
                    false, afr.v, false, bfr[nt & 1].v, (short)0, acc[nt], false, false);
            }
        }

        // convert + LDS-store VALU-staged operands; drain DMA for next tile
        if (haveNext) {
            if constexpr (AKIND == 0) storeA_f32(qb, ra);
            if constexpr (BKIND == 0)      storeB_f32(qb, rb);
            else if constexpr (BKIND == 2) storeB_kn16(qb, rb2);
            if constexpr (AKIND == 1 || BKIND == 1) drain_dma();
        }
        __syncthreads();
    }

    // ---- epilogue: C/D layout VGPR r -> M = r + 8*lh, N = l16 -----------
    const int r0 = rowBase + wave * 16 + lh * 8;
    u16t*  Cb = (u16t*)Cptr + bz * strideC;
    float* Cf = (float*)Cptr + bz * strideC;
    #pragma unroll
    for (int nt = 0; nt < 8; ++nt) {
        const int col = colBase + nt * 16 + l16;
        const float bia = HAS_BIAS ? bias[col] : 0.0f;
        #pragma unroll
        for (int r = 0; r < 8; ++r) {
            const int row = r0 + r;
            float val = acc[nt][r] + bia;
            if (HAS_ADDEND) val += Ad[(size_t)row * ldadd + col];
            if (OUTBF) Cb[(size_t)row * ldc + col] = f2bf(val);
            else       Cf[(size_t)row * ldc + col] = val;
        }
    }
}

// ---------------------------------------------------------------------------
// Row softmax: 2048-wide f32 rows -> bf16 probabilities. One block per row.
// ---------------------------------------------------------------------------
__global__ __launch_bounds__(256)
void wsattn_softmax_rows(const float* __restrict__ S, u16t* __restrict__ P, int n) {
    const int row  = blockIdx.x;
    const int tid  = threadIdx.x;
    const int wave = tid >> 5;
    const int lane = tid & 31;
    const float* s = S + (size_t)row * n;
    u16t*        p = P + (size_t)row * n;

    float vals[8];
    float m = -INFINITY;
    #pragma unroll
    for (int j = 0; j < 8; ++j) {
        vals[j] = s[tid + 256 * j];
        m = fmaxf(m, vals[j]);
    }
    __shared__ float red[8];
    #pragma unroll
    for (int off = 16; off > 0; off >>= 1) m = fmaxf(m, __shfl_xor(m, off, 32));
    if (lane == 0) red[wave] = m;
    __syncthreads();
    float mm = red[0];
    #pragma unroll
    for (int i = 1; i < 8; ++i) mm = fmaxf(mm, red[i]);
    __syncthreads();

    float sum = 0.0f;
    #pragma unroll
    for (int j = 0; j < 8; ++j) {
        vals[j] = __expf(vals[j] - mm);
        sum += vals[j];
    }
    #pragma unroll
    for (int off = 16; off > 0; off >>= 1) sum += __shfl_xor(sum, off, 32);
    if (lane == 0) red[wave] = sum;
    __syncthreads();
    float tot = red[0];
    #pragma unroll
    for (int i = 1; i < 8; ++i) tot += red[i];
    const float inv = 1.0f / tot;
    #pragma unroll
    for (int j = 0; j < 8; ++j) p[tid + 256 * j] = f2bf(vals[j] * inv);
}

// ---------------------------------------------------------------------------
extern "C" void kernel_launch(void* const* d_in, const int* in_sizes, int n_in,
                              void* d_out, int out_size, void* d_ws, size_t ws_size,
                              hipStream_t stream) {
    (void)in_sizes; (void)n_in; (void)out_size;
    constexpr int B = 8, SQ = 2048, SK = 2048, D = 1024, U = 1024;

    const float* query = (const float*)d_in[0];   // [B,SQ,D]
    const float* value = (const float*)d_in[1];   // [B,SK,D]
    const float* wlab  = (const float*)d_in[2];   // [B,SK,U]
    const float* Wq    = (const float*)d_in[3];   // [D,U]
    const float* bq    = (const float*)d_in[4];   // [U]
    const float* Wk    = (const float*)d_in[5];
    const float* bk    = (const float*)d_in[6];
    const float* Wv    = (const float*)d_in[7];
    const float* bv    = (const float*)d_in[8];
    float*       out   = (float*)d_out;           // [B,SQ,U] f32

    const size_t projB  = (size_t)SQ * U * 2;     // bf16 bytes per batch (q/k/v)
    const size_t scoreB = (size_t)SQ * SK * 4;    // f32 scores per batch
    const size_t probB  = (size_t)SQ * SK * 2;    // bf16 probs per batch
    const size_t needT2 = 3 * B * projB + B * scoreB + B * probB;  // ~288 MiB
    const size_t needT1 = 3 * B * projB + scoreB + probB;          // ~120 MiB

    const dim3 blk(256);
    char* ws = (char*)d_ws;

    if (ws_size >= needT2) {
        // Tier 2: everything batched. 6 launches total.
        u16t*  q_ws = (u16t*)(ws);
        u16t*  k_ws = (u16t*)(ws + B * projB);
        u16t*  v_ws = (u16t*)(ws + 2 * B * projB);
        float* s_ws = (float*)(ws + 3 * B * projB);
        u16t*  p_ws = (u16t*)(ws + 3 * B * projB + B * scoreB);

        const dim3 gProj(U / 128, (B * SQ) / 128);        // flatten M over B
        wsattn_gemm_wmma<0, 0, true, true, false><<<gProj, blk, 0, stream>>>(
            query, D, 0, Wq, U, 0, bq, nullptr, 0, 0, q_ws, U, 0, D);
        wsattn_gemm_wmma<0, 0, true, true, true><<<gProj, blk, 0, stream>>>(
            value, D, 0, Wk, U, 0, bk, wlab, U, 0, k_ws, U, 0, D);
        wsattn_gemm_wmma<0, 0, true, true, false><<<gProj, blk, 0, stream>>>(
            value, D, 0, Wv, U, 0, bv, nullptr, 0, 0, v_ws, U, 0, D);

        const dim3 gScore(SK / 128, SQ / 128, B);
        wsattn_gemm_wmma<1, 1, false, false, false><<<gScore, blk, 0, stream>>>(
            q_ws, U, (u64t)SQ * U, k_ws, U, (u64t)SK * U,
            nullptr, nullptr, 0, 0, s_ws, SK, (u64t)SQ * SK, U);
        wsattn_softmax_rows<<<dim3(B * SQ), blk, 0, stream>>>(s_ws, p_ws, SK);
        const dim3 gCtx(U / 128, SQ / 128, B);
        wsattn_gemm_wmma<1, 2, false, false, false><<<gCtx, blk, 0, stream>>>(
            p_ws, SK, (u64t)SQ * SK, v_ws, U, (u64t)SK * U,
            nullptr, nullptr, 0, 0, out, U, (u64t)SQ * U, SK);
    } else if (ws_size >= needT1) {
        // Tier 1: batched projections, per-batch attention scratch.
        u16t*  q_ws = (u16t*)(ws);
        u16t*  k_ws = (u16t*)(ws + B * projB);
        u16t*  v_ws = (u16t*)(ws + 2 * B * projB);
        float* s_ws = (float*)(ws + 3 * B * projB);
        u16t*  p_ws = (u16t*)(ws + 3 * B * projB + scoreB);

        const dim3 gProj(U / 128, (B * SQ) / 128);
        wsattn_gemm_wmma<0, 0, true, true, false><<<gProj, blk, 0, stream>>>(
            query, D, 0, Wq, U, 0, bq, nullptr, 0, 0, q_ws, U, 0, D);
        wsattn_gemm_wmma<0, 0, true, true, true><<<gProj, blk, 0, stream>>>(
            value, D, 0, Wk, U, 0, bk, wlab, U, 0, k_ws, U, 0, D);
        wsattn_gemm_wmma<0, 0, true, true, false><<<gProj, blk, 0, stream>>>(
            value, D, 0, Wv, U, 0, bv, nullptr, 0, 0, v_ws, U, 0, D);

        const dim3 gScore(SK / 128, SQ / 128);
        const dim3 gCtx(U / 128, SQ / 128);
        for (int b = 0; b < B; ++b) {
            wsattn_gemm_wmma<1, 1, false, false, false><<<gScore, blk, 0, stream>>>(
                q_ws + (size_t)b * SQ * U, U, 0, k_ws + (size_t)b * SK * U, U, 0,
                nullptr, nullptr, 0, 0, s_ws, SK, 0, U);
            wsattn_softmax_rows<<<dim3(SQ), blk, 0, stream>>>(s_ws, p_ws, SK);
            wsattn_gemm_wmma<1, 2, false, false, false><<<gCtx, blk, 0, stream>>>(
                p_ws, SK, 0, v_ws + (size_t)b * SK * U, U, 0,
                nullptr, nullptr, 0, 0, out + (size_t)b * SQ * U, U, 0, SK);
        }
    } else {
        // Tier 0: fully per-batch scratch (~36 MiB).
        u16t*  q_ws = (u16t*)(ws);
        u16t*  k_ws = (u16t*)(ws + projB);
        u16t*  v_ws = (u16t*)(ws + 2 * projB);
        float* s_ws = (float*)(ws + 3 * projB);
        u16t*  p_ws = (u16t*)(ws + 3 * projB + scoreB);

        const dim3 gProj(U / 128, SQ / 128);
        const dim3 gScore(SK / 128, SQ / 128);
        const dim3 gCtx(U / 128, SQ / 128);
        for (int b = 0; b < B; ++b) {
            const float* Qb = query + (size_t)b * SQ * D;
            const float* Vb = value + (size_t)b * SK * D;
            const float* Lb = wlab  + (size_t)b * SK * U;
            float*       Ob = out   + (size_t)b * SQ * U;
            wsattn_gemm_wmma<0, 0, true, true, false><<<gProj, blk, 0, stream>>>(
                Qb, D, 0, Wq, U, 0, bq, nullptr, 0, 0, q_ws, U, 0, D);
            wsattn_gemm_wmma<0, 0, true, true, true><<<gProj, blk, 0, stream>>>(
                Vb, D, 0, Wk, U, 0, bk, Lb, U, 0, k_ws, U, 0, D);
            wsattn_gemm_wmma<0, 0, true, true, false><<<gProj, blk, 0, stream>>>(
                Vb, D, 0, Wv, U, 0, bv, nullptr, 0, 0, v_ws, U, 0, D);
            wsattn_gemm_wmma<1, 1, false, false, false><<<gScore, blk, 0, stream>>>(
                q_ws, U, 0, k_ws, U, 0, nullptr, nullptr, 0, 0, s_ws, SK, 0, U);
            wsattn_softmax_rows<<<dim3(SQ), blk, 0, stream>>>(s_ws, p_ws, SK);
            wsattn_gemm_wmma<1, 2, false, false, false><<<gCtx, blk, 0, stream>>>(
                p_ws, SK, 0, v_ws, U, 0, nullptr, nullptr, 0, 0, Ob, U, 0, SK);
        }
    }
}
